// up_fcn_3x3_s4_dilationHead_90297392431651
// MI455X (gfx1250) — matrix operand order repaired and checked by reading, not compiled, not run on hardware
//
#include <hip/hip_runtime.h>
#include <hip/hip_bf16.h>

// CDNA5 wave32 WMMA types
typedef __bf16  v16bf __attribute__((ext_vector_type(16)));
typedef float   v8f   __attribute__((ext_vector_type(8)));

#define BN_EPS 1e-5f

// ---------------------------------------------------------------------------
// Generic 1x1-conv GEMM via V_WMMA_F32_16X16X32_BF16.
// Y[n][m][p] = sum_k W[m][k] * X[n][k][p]  (+ optional BN / bias / relu)
// Block = 256 threads = 8 waves. Each wave computes a 16(outch) x 32(pixel)
// tile as two WMMAs sharing one A fragment; the 32px x 32k activation tile is
// staged in LDS as bf16, [pixel][k], read in contiguous k order (k = 16h + s) as A is.
// ---------------------------------------------------------------------------
__global__ void gemm1x1_wmma(const float* __restrict__ X,
                             const float* __restrict__ Wt,
                             float* __restrict__ Y,
                             const float* __restrict__ bn,    // (4,Cout) or null
                             const float* __restrict__ bias,  // (Cout)  or null
                             int Cin, int Cout, int P, int relu)
{
    __shared__ __align__(32) __bf16 Bs[32 * 32];   // [pixel][k]
    const int tid   = threadIdx.x;
    const int wave  = tid >> 5;
    const int lane  = tid & 31;
    const int p_base = blockIdx.x * 32;
    const int m_tile = blockIdx.y * 128 + wave * 16;
    const int n      = blockIdx.z;

    const float* Xn = X + (size_t)n * Cin * P;
    const int  m_row = m_tile + (lane & 15);
    const bool mval  = (m_row < Cout);
    const float* wrow = Wt + (size_t)(mval ? m_row : 0) * Cin;
    const int kbA = (lane < 16) ? 0 : 16;   // A-matrix K base per half-wave
    const int kbB = (lane < 16) ? 0 : 16;   // B-matrix K base per half-wave
    const int npx = lane & 15;

    v8f acc0 = {}, acc1 = {};
    for (int k0 = 0; k0 < Cin; k0 += 32) {
        // cooperatively stage activation tile (bf16): Bs[px][kk], fully unrolled
#pragma unroll
        for (int it = 0; it < 4; ++it) {
            int e  = tid + it * 256;          // 0..1023
            int px = e >> 5, kk = e & 31;
            Bs[px * 32 + kk] = (__bf16)Xn[(size_t)(k0 + kk) * P + p_base + px];
        }
        __syncthreads();

        // A: lanes 0-15 -> K {0..15}; lanes 16-31 -> K {16..31}
        v16bf a;
        if (mval) {
            const float* wp = wrow + k0 + kbA;
#pragma unroll
            for (int j = 0; j < 8; ++j) a[j]     = (__bf16)wp[j];
#pragma unroll
            for (int j = 0; j < 8; ++j) a[8 + j] = (__bf16)wp[8 + j];
            if (k0 + 32 < Cin) __builtin_prefetch(wrow + k0 + 32, 0, 1);
        } else {
#pragma unroll
            for (int j = 0; j < 16; ++j) a[j] = (__bf16)0.0f;
        }
        // B: lane's pixel column, 16 contiguous K values; two pixel sub-tiles
        v16bf b0 = *(const v16bf*)&Bs[npx * 32 + kbB];
        v16bf b1 = *(const v16bf*)&Bs[(npx + 16) * 32 + kbB];

        acc0 = __builtin_amdgcn_wmma_f32_16x16x32_bf16(
            false, a, false, b0, (short)0, acc0, false, false);
        acc1 = __builtin_amdgcn_wmma_f32_16x16x32_bf16(
            false, a, false, b1, (short)0, acc1, false, false);
        __syncthreads();
    }

    // epilogue: C/D layout — VGPR r holds M=r (lanes 0-15) / M=r+8 (lanes 16-31)
#pragma unroll
    for (int r = 0; r < 8; ++r) {
        int m = m_tile + r + ((lane < 16) ? 0 : 8);
        if (m >= Cout) continue;
        float v0 = acc0[r], v1 = acc1[r];
        if (bn) {
            float scale = bn[m] * __frsqrt_rn(bn[3 * Cout + m] + BN_EPS);
            float shift = bn[Cout + m] - bn[2 * Cout + m] * scale;
            v0 = v0 * scale + shift;
            v1 = v1 * scale + shift;
        }
        if (bias) { v0 += bias[m]; v1 += bias[m]; }
        if (relu) { v0 = fmaxf(v0, 0.0f); v1 = fmaxf(v1, 0.0f); }
        float* yr = Y + ((size_t)n * Cout + m) * P + p_base + npx;
        yr[0]  = v0;
        yr[16] = v1;
    }
}

// ---------------------------------------------------------------------------
// conv5: 3x3 conv (pad 1) as implicit GEMM, K = Cin*9, fused BN + ReLU.
// Same 16x32 wave tile as gemm1x1.
// ---------------------------------------------------------------------------
__global__ void conv3x3_bn_relu_wmma(const float* __restrict__ X,
                                     const float* __restrict__ Wt,  // (O,Cin,3,3)
                                     float* __restrict__ Y,
                                     const float* __restrict__ bn,  // (4,Cout)
                                     int Cin, int Cout, int H, int Wd)
{
    __shared__ __align__(32) __bf16 Bs[32 * 32];
    const int P = H * Wd;
    const int K = Cin * 9;
    const int tid  = threadIdx.x;
    const int wave = tid >> 5;
    const int lane = tid & 31;
    const int p_base = blockIdx.x * 32;
    const int m_tile = blockIdx.y * 128 + wave * 16;
    const int n      = blockIdx.z;

    const float* Xn = X + (size_t)n * Cin * P;
    const int  m_row = m_tile + (lane & 15);
    const bool mval  = (m_row < Cout);
    const float* wrow = Wt + (size_t)(mval ? m_row : 0) * K;
    const int kbA = (lane < 16) ? 0 : 16;
    const int kbB = (lane < 16) ? 0 : 16;
    const int npx = lane & 15;

    v8f acc0 = {}, acc1 = {};
    for (int k0 = 0; k0 < K; k0 += 32) {
#pragma unroll
        for (int it = 0; it < 4; ++it) {
            int e  = tid + it * 256;
            int px = e >> 5, kk = e & 31;
            int k  = k0 + kk;
            int c  = k / 9, r = k - c * 9;
            int di = r / 3 - 1;
            int dj = r - (r / 3) * 3 - 1;
            int p  = p_base + px;
            int y  = p / Wd + di;
            int x  = p - (p / Wd) * Wd + dj;
            float v = (y >= 0 && y < H && x >= 0 && x < Wd)
                          ? Xn[(size_t)c * P + y * Wd + x] : 0.0f;
            Bs[px * 32 + kk] = (__bf16)v;
        }
        __syncthreads();

        v16bf a;
        if (mval) {
            const float* wp = wrow + k0 + kbA;
#pragma unroll
            for (int j = 0; j < 8; ++j) a[j]     = (__bf16)wp[j];
#pragma unroll
            for (int j = 0; j < 8; ++j) a[8 + j] = (__bf16)wp[8 + j];
            if (k0 + 32 < K) __builtin_prefetch(wrow + k0 + 32, 0, 1);
        } else {
#pragma unroll
            for (int j = 0; j < 16; ++j) a[j] = (__bf16)0.0f;
        }
        v16bf b0 = *(const v16bf*)&Bs[npx * 32 + kbB];
        v16bf b1 = *(const v16bf*)&Bs[(npx + 16) * 32 + kbB];

        acc0 = __builtin_amdgcn_wmma_f32_16x16x32_bf16(
            false, a, false, b0, (short)0, acc0, false, false);
        acc1 = __builtin_amdgcn_wmma_f32_16x16x32_bf16(
            false, a, false, b1, (short)0, acc1, false, false);
        __syncthreads();
    }

#pragma unroll
    for (int r = 0; r < 8; ++r) {
        int m = m_tile + r + ((lane < 16) ? 0 : 8);
        if (m >= Cout) continue;
        float scale = bn[m] * __frsqrt_rn(bn[3 * Cout + m] + BN_EPS);
        float shift = bn[Cout + m] - bn[2 * Cout + m] * scale;
        float* yr = Y + ((size_t)n * Cout + m) * P + p_base + npx;
        yr[0]  = fmaxf(acc0[r] * scale + shift, 0.0f);
        yr[16] = fmaxf(acc1[r] * scale + shift, 0.0f);
    }
}

// ---------------------------------------------------------------------------
// bilinear 2x upsample, align_corners=True  (in: BC x h x w -> out: BC x 2h x 2w)
// ---------------------------------------------------------------------------
__global__ void upsample2x_ac(const float* __restrict__ in, float* __restrict__ out,
                              int BC, int h, int w)
{
    const int H = h * 2, Wd = w * 2;
    const float sy = (H > 1) ? (float)(h - 1) / (float)(H - 1) : 0.0f;
    const float sx = (Wd > 1) ? (float)(w - 1) / (float)(Wd - 1) : 0.0f;
    const size_t total = (size_t)BC * H * Wd;
    const size_t stride = (size_t)gridDim.x * blockDim.x;
    for (size_t i = (size_t)blockIdx.x * blockDim.x + threadIdx.x; i < total; i += stride) {
        int x  = (int)(i % Wd);
        int y  = (int)((i / Wd) % H);
        size_t bc = i / ((size_t)H * Wd);
        float fy = y * sy, fx = x * sx;
        int y0 = (int)fy, x0 = (int)fx;
        int y1 = min(y0 + 1, h - 1), x1 = min(x0 + 1, w - 1);
        float wy = fy - y0, wx = fx - x0;
        const float* p = in + bc * (size_t)h * w;
        float v00 = p[y0 * w + x0], v01 = p[y0 * w + x1];
        float v10 = p[y1 * w + x0], v11 = p[y1 * w + x1];
        out[i] = (v00 * (1.f - wx) + v01 * wx) * (1.f - wy) +
                 (v10 * (1.f - wx) + v11 * wx) * wy;
    }
}

// ---------------------------------------------------------------------------
// 9-tap dilated (d=2) local attention: att[n][j][p] = softmax_j( q . unfold9(k) )
// ---------------------------------------------------------------------------
__global__ void local_attn9(const float* __restrict__ q, const float* __restrict__ k,
                            float* __restrict__ att, int kd, int H, int Wd, int B)
{
    const int P = H * Wd;
    int idx = blockIdx.x * blockDim.x + threadIdx.x;
    if (idx >= B * P) return;
    int n = idx / P, p = idx - n * P;
    int y = p / Wd, x = p - (p / Wd) * Wd;
    const float* qn = q + (size_t)n * kd * P;
    const float* kn = k + (size_t)n * kd * P;
    float acc[9];
#pragma unroll
    for (int j = 0; j < 9; ++j) acc[j] = 0.0f;
    for (int c = 0; c < kd; ++c) {
        float qv = qn[(size_t)c * P + p];
        const float* kc = kn + (size_t)c * P;
#pragma unroll
        for (int j = 0; j < 9; ++j) {
            int yy = y + (j / 3) * 2 - 2;
            int xx = x + (j % 3) * 2 - 2;
            float kv = (yy >= 0 && yy < H && xx >= 0 && xx < Wd) ? kc[yy * Wd + xx] : 0.0f;
            acc[j] += qv * kv;
        }
    }
    float mx = acc[0];
#pragma unroll
    for (int j = 1; j < 9; ++j) mx = fmaxf(mx, acc[j]);
    float s = 0.0f;
#pragma unroll
    for (int j = 0; j < 9; ++j) { acc[j] = __expf(acc[j] - mx); s += acc[j]; }
    float inv = 1.0f / s;
#pragma unroll
    for (int j = 0; j < 9; ++j)
        att[((size_t)n * 9 + j) * P + p] = acc[j] * inv;
}

// ---------------------------------------------------------------------------
// out[n][c][p] = sum_j att[n][j][p] * unfold9(v)[n][c][j][p]
// ---------------------------------------------------------------------------
__global__ void local_agg9(const float* __restrict__ att, const float* __restrict__ v,
                           float* __restrict__ out, int C, int H, int Wd, int B)
{
    const int P = H * Wd;
    const size_t total  = (size_t)B * C * P;
    const size_t stride = (size_t)gridDim.x * blockDim.x;
    for (size_t i = (size_t)blockIdx.x * blockDim.x + threadIdx.x; i < total; i += stride) {
        int p = (int)(i % P);
        size_t nc = i / P;
        int c = (int)(nc % C);
        int n = (int)(nc / C);
        int y = p / Wd, x = p - (p / Wd) * Wd;
        const float* an = att + (size_t)n * 9 * P;
        const float* vc = v + ((size_t)n * C + c) * P;
        float s = 0.0f;
#pragma unroll
        for (int j = 0; j < 9; ++j) {
            int yy = y + (j / 3) * 2 - 2;
            int xx = x + (j % 3) * 2 - 2;
            if (yy >= 0 && yy < H && xx >= 0 && xx < Wd)
                s += an[(size_t)j * P + p] * vc[yy * Wd + xx];
        }
        out[i] = s;
    }
}

// ---------------------------------------------------------------------------
extern "C" void kernel_launch(void* const* d_in, const int* in_sizes, int n_in,
                              void* d_out, int out_size, void* d_ws, size_t ws_size,
                              hipStream_t stream)
{
    const float* c1      = (const float*)d_in[0];   // (2,256,128,128)
    const float* c2      = (const float*)d_in[1];   // (2,512,64,64)
    const float* c3      = (const float*)d_in[2];   // (2,1024,32,32)
    const float* c4      = (const float*)d_in[3];   // (2,2048,16,16)
    /* c20 = d_in[4] unused by the reference math */
    const float* c30     = (const float*)d_in[5];   // (2,1024,32,32)
    const float* c40     = (const float*)d_in[6];   // (2,2048,16,16)
    const float* conv5_w = (const float*)d_in[7];   // (512,2048,3,3)
    const float* bn5     = (const float*)d_in[8];
    const float* conv6_w = (const float*)d_in[9];   // (19,512)
    const float* conv6_b = (const float*)d_in[10];
    const float* lu4_w1  = (const float*)d_in[11];  const float* lu4_bn1 = (const float*)d_in[12];
    const float* lu4_w2  = (const float*)d_in[13];  const float* lu4_bn2 = (const float*)d_in[14];
    const float* lu3_w1  = (const float*)d_in[15];  const float* lu3_bn1 = (const float*)d_in[16];
    const float* lu3_w2  = (const float*)d_in[17];  const float* lu3_bn2 = (const float*)d_in[18];
    const float* lu2_w1  = (const float*)d_in[19];  const float* lu2_bn1 = (const float*)d_in[20];
    const float* lu2_w2  = (const float*)d_in[21];  const float* lu2_bn2 = (const float*)d_in[22];

    const int B = 2;
    // workspace layout (floats): two 67MB regions reused across stages
    const size_t NBIG = (size_t)B * 512 * 128 * 128;   // 16,777,216 floats
    float* U1  = (float*)d_ws;          // upsample scratch (c_lo_up then out_prev_up)
    float* OUT = U1 + NBIG;             // running "out" tensor (overwritten in place)
    float* Qb  = OUT + NBIG;            // q projection (max 2*64*16384)
    float* Kb  = Qb + 2097152;          // k projection
    float* ATT = Kb + 2097152;          // attention (max 2*9*16384)

    auto ups = [&](const float* in, float* out, int BC, int h, int w) {
        size_t total = (size_t)BC * h * 2 * w * 2;
        int blocks = (int)((total + 255) / 256);
        if (blocks > 16384) blocks = 16384;
        upsample2x_ac<<<blocks, 256, 0, stream>>>(in, out, BC, h, w);
    };
    auto agg = [&](const float* att, const float* v, float* out, int C, int H, int W) {
        size_t total = (size_t)B * C * H * W;
        int blocks = (int)((total + 255) / 256);
        if (blocks > 65536) blocks = 65536;
        local_agg9<<<blocks, 256, 0, stream>>>(att, v, out, C, H, W, B);
    };

    // ---- conv5: 3x3 2048->512 on 16x16, BN+ReLU -> OUT (2,512,16,16)
    conv3x3_bn_relu_wmma<<<dim3(8, 4, B), 256, 0, stream>>>(
        c4, conv5_w, OUT, bn5, 2048, 512, 16, 16);

    // ---- localUp4: c_hi=c3 (1024,32x32), c_lo=c40 (2048,16x16), kd=128
    gemm1x1_wmma<<<dim3(32, 1, B), 256, 0, stream>>>(c3, lu4_w1, Qb, lu4_bn1, nullptr, 1024, 128, 1024, 0);
    ups(c40, U1, B * 2048, 16, 16);
    gemm1x1_wmma<<<dim3(32, 1, B), 256, 0, stream>>>(U1, lu4_w2, Kb, lu4_bn2, nullptr, 2048, 128, 1024, 0);
    ups(OUT, U1, B * 512, 16, 16);
    local_attn9<<<(B * 1024 + 255) / 256, 256, 0, stream>>>(Qb, Kb, ATT, 128, 32, 32, B);
    agg(ATT, U1, OUT, 512, 32, 32);                 // OUT := (2,512,32,32)

    // ---- localUp3: c_hi=c2 (512,64x64), c_lo=c30 (1024,32x32), kd=64
    gemm1x1_wmma<<<dim3(128, 1, B), 256, 0, stream>>>(c2, lu3_w1, Qb, lu3_bn1, nullptr, 512, 64, 4096, 0);
    ups(c30, U1, B * 1024, 32, 32);
    gemm1x1_wmma<<<dim3(128, 1, B), 256, 0, stream>>>(U1, lu3_w2, Kb, lu3_bn2, nullptr, 1024, 64, 4096, 0);
    ups(OUT, U1, B * 512, 32, 32);
    local_attn9<<<(B * 4096 + 255) / 256, 256, 0, stream>>>(Qb, Kb, ATT, 64, 64, 64, B);
    agg(ATT, U1, OUT, 512, 64, 64);                 // OUT := (2,512,64,64)

    // ---- localUp2: c_hi=c1 (256,128x128), c_lo=c2 (512,64x64), kd=64
    gemm1x1_wmma<<<dim3(512, 1, B), 256, 0, stream>>>(c1, lu2_w1, Qb, lu2_bn1, nullptr, 256, 64, 16384, 0);
    ups(c2, U1, B * 512, 64, 64);
    gemm1x1_wmma<<<dim3(512, 1, B), 256, 0, stream>>>(U1, lu2_w2, Kb, lu2_bn2, nullptr, 512, 64, 16384, 0);
    ups(OUT, U1, B * 512, 64, 64);
    local_attn9<<<(B * 16384 + 255) / 256, 256, 0, stream>>>(Qb, Kb, ATT, 64, 128, 128, B);
    agg(ATT, U1, OUT, 512, 128, 128);               // OUT := (2,512,128,128)

    // ---- conv6: 1x1 512->19 + bias -> d_out (2,19,128,128)
    gemm1x1_wmma<<<dim3(512, 1, B), 256, 0, stream>>>(
        OUT, conv6_w, (float*)d_out, nullptr, conv6_b, 512, 19, 16384, 0);
}
